// MultiHeadedCacheAttention_49194555408846
// MI455X (gfx1250) — matrix operand order repaired and checked by reading, not compiled, not run on hardware
//
#include <hip/hip_runtime.h>
#include <hip/hip_bf16.h>

#define B_   2
#define S_   2048
#define DIM_ 1024
#define H_   16
#define HK_  4
#define D_   64

typedef __attribute__((ext_vector_type(16))) __bf16 v16bf;
typedef __attribute__((ext_vector_type(8)))  float  v8f;
typedef __attribute__((ext_vector_type(4)))  unsigned int v4u32;
typedef __attribute__((ext_vector_type(8)))  int  v8i32;
typedef __attribute__((ext_vector_type(4)))  int  v4i32;

#if __has_builtin(__builtin_amdgcn_tensor_load_to_lds) && \
    __has_builtin(__builtin_amdgcn_s_wait_tensorcnt)
#define HAVE_TDM 1
#else
#define HAVE_TDM 0
#endif

__device__ __forceinline__ unsigned short f2bf(float f) {
  unsigned u = __float_as_uint(f);
  u += 0x7FFFu + ((u >> 16) & 1u);          // round to nearest even
  return (unsigned short)(u >> 16);
}

union BF16Frag { v16bf v; unsigned u[8]; };

// ---------------------------------------------------------------------------
// GEMM: C[M,N] = A[M,K] (fp32 or bf16) * W[K,N] (fp32 -> bf16) + bias
// Block: 256 threads = 8 waves. Block tile 64(M) x 64(N); wave tile 16x32
// (two V_WMMA_F32_16X16X32_BF16 per wave per K-step, A fragment reused).
// ---------------------------------------------------------------------------
template<bool A_BF16, bool OUT_BF16>
__global__ __launch_bounds__(256)
void gemm_bf16_wmma(const void* __restrict__ Aptr,
                    const float* __restrict__ W,
                    const float* __restrict__ bias,
                    void* __restrict__ Cptr,
                    int M, int N, int K)
{
  __shared__ unsigned short As[64 * 32];   // [m][k]  (pairs along k)
  __shared__ unsigned short Ws[64 * 32];   // [n][k]  transposed (pairs along k)

  const int tid  = threadIdx.x;
  const int lane = tid & 31;
  const int wave = tid >> 5;
  const int h    = lane >> 4;              // lane half (wave32)
  const int l15  = lane & 15;

  const int m0 = blockIdx.y * 64;
  const int n0 = blockIdx.x * 64;
  const int mbase = (wave >> 1) * 16;      // 4 wave-rows
  const int nbase = (wave & 1) * 32;       // 2 wave-cols x 32

  v8f acc0 = {}, acc1 = {};

  for (int kk = 0; kk < K; kk += 32) {
    // ---- stage A tile 64x32 (8 elems / thread) ----
    {
      int r  = tid >> 2;
      int ks = (tid & 3) * 8;
      if constexpr (A_BF16) {
        const unsigned short* A = (const unsigned short*)Aptr;
        uint4 d0 = *(const uint4*)(A + (size_t)(m0 + r) * K + kk + ks);
        *(uint4*)&As[r * 32 + ks] = d0;
        if (kk + 32 < K)
          __builtin_prefetch(A + (size_t)(m0 + r) * K + kk + 32 + ks, 0, 0);
      } else {
        const float* A = (const float*)Aptr;
        const float4* src = (const float4*)(A + (size_t)(m0 + r) * K + kk + ks);
        float4 f0 = src[0], f1 = src[1];
        union { unsigned short s[8]; uint4 q; } t;
        t.s[0] = f2bf(f0.x); t.s[1] = f2bf(f0.y); t.s[2] = f2bf(f0.z); t.s[3] = f2bf(f0.w);
        t.s[4] = f2bf(f1.x); t.s[5] = f2bf(f1.y); t.s[6] = f2bf(f1.z); t.s[7] = f2bf(f1.w);
        *(uint4*)&As[r * 32 + ks] = t.q;
        if (kk + 32 < K)
          __builtin_prefetch(A + (size_t)(m0 + r) * K + kk + 32 + ks, 0, 0);
      }
    }
    // ---- stage W tile 32x64, transposed into [n][k] (8 elems / thread) ----
    {
      int kr = tid >> 3;
      int ns = (tid & 7) * 8;
      const float* Wp = W + (size_t)(kk + kr) * N + n0 + ns;
      float4 f0 = *(const float4*)(Wp);
      float4 f1 = *(const float4*)(Wp + 4);
      Ws[(ns + 0) * 32 + kr] = f2bf(f0.x);
      Ws[(ns + 1) * 32 + kr] = f2bf(f0.y);
      Ws[(ns + 2) * 32 + kr] = f2bf(f0.z);
      Ws[(ns + 3) * 32 + kr] = f2bf(f0.w);
      Ws[(ns + 4) * 32 + kr] = f2bf(f1.x);
      Ws[(ns + 5) * 32 + kr] = f2bf(f1.y);
      Ws[(ns + 6) * 32 + kr] = f2bf(f1.z);
      Ws[(ns + 7) * 32 + kr] = f2bf(f1.w);
    }
    __syncthreads();

    // A fragment: lane row = l15; K blocks of 8 split across lane halves
    BF16Frag a, b0, b1;
#pragma unroll
    for (int v = 0; v < 8; ++v) {
      int kp = (v < 4) ? (h * 8 + 2 * v) : (16 + h * 8 + 2 * (v - 4));
      a.u[v] = *(const unsigned*)&As[(mbase + l15) * 32 + kp];
    }
    // B fragments: lane col = l15; K blocks of 8 split across lane halves
#pragma unroll
    for (int v = 0; v < 8; ++v) {
      int kp = (v < 4) ? (h * 8 + 2 * v) : (16 + h * 8 + 2 * (v - 4));
      b0.u[v] = *(const unsigned*)&Ws[(nbase + l15) * 32 + kp];
      b1.u[v] = *(const unsigned*)&Ws[(nbase + 16 + l15) * 32 + kp];
    }
    acc0 = __builtin_amdgcn_wmma_f32_16x16x32_bf16(false, a.v, false, b0.v,
                                                   (short)0, acc0, false, false);
    acc1 = __builtin_amdgcn_wmma_f32_16x16x32_bf16(false, a.v, false, b1.v,
                                                   (short)0, acc1, false, false);
    __syncthreads();
  }

  // epilogue: C/D layout -> M = r + 8*h, N = l15
#pragma unroll
  for (int r = 0; r < 8; ++r) {
    int m  = m0 + mbase + r + 8 * h;
    int n  = n0 + nbase + l15;
    int n2 = n + 16;
    float v0 = acc0[r] + bias[n];
    float v1 = acc1[r] + bias[n2];
    if constexpr (OUT_BF16) {
      ((unsigned short*)Cptr)[(size_t)m * N + n]  = f2bf(v0);
      ((unsigned short*)Cptr)[(size_t)m * N + n2] = f2bf(v1);
    } else {
      ((float*)Cptr)[(size_t)m * N + n]  = v0;
      ((float*)Cptr)[(size_t)m * N + n2] = v1;
    }
  }
}

// ---------------------------------------------------------------------------
// Fused flash attention (GQA). Block: 128 threads = 4 waves, one (b,h) per
// block, each wave owns 16 q-rows x D=64. Waves share staged K/V chunks.
// K chunk is staged by the Tensor Data Mover (2D descriptor load to LDS).
// ---------------------------------------------------------------------------
__global__ __launch_bounds__(128)
void attn_wmma(const unsigned short* __restrict__ qb,
               const unsigned short* __restrict__ kb,
               const unsigned short* __restrict__ vb,
               const int* __restrict__ mask_x,
               const int* __restrict__ mask_y,
               const int* __restrict__ attn_mask,
               unsigned short* __restrict__ hb)
{
  __shared__ unsigned short kT[32 * 64];     // [key][d]   (pairs along d)
  __shared__ unsigned short vT[64 * 32];     // [d][key]   (pairs along key)
  __shared__ unsigned short pT[4][16 * 32];  // per-wave probs [m][k-in-chunk]

  const int tid  = threadIdx.x;
  const int lane = tid & 31;
  const int wave = tid >> 5;
  const int h    = lane >> 4;
  const int l15  = lane & 15;

  const int b  = blockIdx.z;
  const int hh = blockIdx.y;                  // query head
  const int hk = hh / (H_ / HK_);             // kv head
  const int q0 = blockIdx.x * 64 + wave * 16; // this wave's q-tile

  // ---- persistent Q fragments: two K=32 tiles covering D=64 ----
  BF16Frag qf[2];
#pragma unroll
  for (int t = 0; t < 2; ++t)
#pragma unroll
    for (int v = 0; v < 8; ++v) {
      int kp = (v < 4) ? (h * 8 + 2 * v) : (16 + h * 8 + 2 * (v - 4));
      int d  = t * 32 + kp;
      qf[t].u[v] = *(const unsigned*)
          &qb[((size_t)(b * S_ + q0 + l15) * H_ + hh) * D_ + d];
    }

  float mx[8];
#pragma unroll
  for (int r = 0; r < 8; ++r)
    mx[r] = (float)mask_x[b * S_ + q0 + r + 8 * h];

  float m_run[8], s_run[8];
#pragma unroll
  for (int r = 0; r < 8; ++r) { m_run[r] = -3.0e38f; s_run[r] = 0.0f; }
  v8f o[4] = {{}, {}, {}, {}};               // 16 x 64 fp32 accumulators

  for (int kc = 0; kc < S_ / 32; ++kc) {
    const int ks0 = kc * 32;

#if HAVE_TDM
    // ---- K chunk via Tensor Data Mover: 2D tile 32(keys) x 64(d), bf16,
    //      row stride HK_*D_ elements, straight into kT[key][d]. Wave 0 only
    //      (TDM is a per-wave DMA; EXEC ignored). ----
    if (wave == 0) {
      unsigned long long ga = (unsigned long long)(uintptr_t)
          (kb + ((size_t)(b * S_ + ks0) * HK_ + hk) * D_);
      unsigned ldsa = (unsigned)(uintptr_t)&kT[0];   // LDS byte offset
      v4u32 g0 = { 1u,                                // count=1, user mode
                   ldsa,                              // lds_addr
                   (unsigned)(ga & 0xFFFFFFFFu),      // global_addr[31:0]
                   (unsigned)((ga >> 32) & 0x01FFFFFFu) | (2u << 30) }; // type=2
      v8i32 g1 = { (int)(1u << 16),      // data_size = 1 (2 bytes)
                   (int)(64u << 16),     // tensor_dim0[15:0] = 64
                   (int)(32u << 16),     // tensor_dim0 hi=0 | tensor_dim1[15:0]=32
                   (int)(64u << 16),     // tensor_dim1 hi=0 | tile_dim0=64
                   (int)(32u),           // tile_dim1=32, tile_dim2=0
                   (int)(HK_ * D_),      // tensor_dim0_stride = 256 elems
                   0, 0 };
      v4i32 gz4 = { 0, 0, 0, 0 };
      v8i32 gz8 = { 0, 0, 0, 0, 0, 0, 0, 0 };
      __builtin_amdgcn_tensor_load_to_lds(g0, g1, gz4, gz4, gz8, 0);
    }
#endif
    // ---- V chunk staged manually (needs [d][key] transpose) ----
    {
      int key = tid >> 2;                    // 0..31
      int ds  = (tid & 3) * 16;              // 0,16,32,48
      const uint4* vsrc = (const uint4*)
          &vb[((size_t)(b * S_ + ks0 + key) * HK_ + hk) * D_ + ds];
      union { uint4 q[2]; unsigned short s[16]; } vv;
      vv.q[0] = vsrc[0]; vv.q[1] = vsrc[1];
#pragma unroll
      for (int j = 0; j < 16; ++j)           // transpose into [d][key]
        vT[(ds + j) * 32 + key] = vv.s[j];
#if !HAVE_TDM
      const uint4* ksrc = (const uint4*)
          &kb[((size_t)(b * S_ + ks0 + key) * HK_ + hk) * D_ + ds];
      uint4 k0 = ksrc[0], k1 = ksrc[1];
      *(uint4*)&kT[key * 64 + ds]     = k0;
      *(uint4*)&kT[key * 64 + ds + 8] = k1;
#endif
    }
    // prefetch next chunk's attn_mask rows while DMA/staging is in flight
    if (kc + 1 < S_ / 32)
      __builtin_prefetch(&attn_mask[(size_t)(b * S_ + q0 + l15) * S_ + ks0 + 32], 0, 0);
#if HAVE_TDM
    __builtin_amdgcn_s_wait_tensorcnt(0);
#endif
    __syncthreads();

    // ---- scores = Q . K^T for 2 subtiles of 16 keys (contraction D=64) ----
    float sA[8], sB[8];
#pragma unroll
    for (int st = 0; st < 2; ++st) {
      v8f sc = {};
#pragma unroll
      for (int t = 0; t < 2; ++t) {
        BF16Frag bk;
#pragma unroll
        for (int v = 0; v < 8; ++v) {
          int d = t * 32 + ((v < 4) ? (h * 8 + 2 * v) : (16 + h * 8 + 2 * (v - 4)));   // B-frag: N=key col, K=d
          bk.u[v] = *(const unsigned*)&kT[(st * 16 + l15) * 64 + d];
        }
        sc = __builtin_amdgcn_wmma_f32_16x16x32_bf16(false, qf[t].v, false, bk.v,
                                                     (short)0, sc, false, false);
      }
      int key = ks0 + st * 16 + l15;
      float my = (float)mask_y[b * S_ + key];
#pragma unroll
      for (int r = 0; r < 8; ++r) {
        float am = (float)attn_mask[(size_t)(b * S_ + q0 + r + 8 * h) * S_ + key];
        float s = sc[r] * 0.125f                     // 1/sqrt(64)
                  - 100000.0f * (1.0f - mx[r] * my)
                  - 100000.0f * (1.0f - am);
        if (st == 0) sA[r] = s; else sB[r] = s;
      }
    }

    // ---- online softmax (row reductions within 16-lane halves) ----
    float fac[8];
#pragma unroll
    for (int r = 0; r < 8; ++r) {
      float cm = fmaxf(sA[r], sB[r]);
      cm = fmaxf(cm, __shfl_xor(cm, 1, 32));
      cm = fmaxf(cm, __shfl_xor(cm, 2, 32));
      cm = fmaxf(cm, __shfl_xor(cm, 4, 32));
      cm = fmaxf(cm, __shfl_xor(cm, 8, 32));
      float mn = fmaxf(m_run[r], cm);
      fac[r] = __expf(m_run[r] - mn);
      m_run[r] = mn;
      sA[r] = __expf(sA[r] - mn);
      sB[r] = __expf(sB[r] - mn);
      float rs = sA[r] + sB[r];
      rs += __shfl_xor(rs, 1, 32);
      rs += __shfl_xor(rs, 2, 32);
      rs += __shfl_xor(rs, 4, 32);
      rs += __shfl_xor(rs, 8, 32);
      s_run[r] = s_run[r] * fac[r] + rs;
    }
#pragma unroll
    for (int t4 = 0; t4 < 4; ++t4)
#pragma unroll
      for (int r = 0; r < 8; ++r)
        o[t4][r] *= fac[r];

    // ---- probs -> per-wave LDS tile, reload in A-fragment layout ----
#pragma unroll
    for (int r = 0; r < 8; ++r) {
      pT[wave][(r + 8 * h) * 32 + l15]      = f2bf(sA[r]);
      pT[wave][(r + 8 * h) * 32 + 16 + l15] = f2bf(sB[r]);
    }
    asm volatile("s_wait_dscnt 0" ::: "memory");   // wave-internal LDS RAW
    BF16Frag pf;
#pragma unroll
    for (int v = 0; v < 8; ++v) {
      int kp = (v < 4) ? (h * 8 + 2 * v) : (16 + h * 8 + 2 * (v - 4));
      pf.u[v] = *(const unsigned*)&pT[wave][l15 * 32 + kp];
    }

    // ---- o[16x64] += P[16x32] * V[32x64]  (4 WMMA) ----
#pragma unroll
    for (int t4 = 0; t4 < 4; ++t4) {
      BF16Frag bvf;
#pragma unroll
      for (int v = 0; v < 8; ++v) {
        int key = (v < 4) ? (h * 8 + 2 * v) : (16 + h * 8 + 2 * (v - 4));            // B-frag: N=d col, K=key
        bvf.u[v] = *(const unsigned*)&vT[(t4 * 16 + l15) * 32 + key];
      }
      o[t4] = __builtin_amdgcn_wmma_f32_16x16x32_bf16(false, pf.v, false, bvf.v,
                                                      (short)0, o[t4], false, false);
    }
    __syncthreads();
  }

  // ---- normalize and write bf16 context ----
#pragma unroll
  for (int t4 = 0; t4 < 4; ++t4)
#pragma unroll
    for (int r = 0; r < 8; ++r) {
      int q = q0 + r + 8 * h;
      int d = t4 * 16 + l15;
      float val = o[t4][r] / s_run[r];
      hb[(size_t)(b * S_ + q) * (H_ * D_) + hh * D_ + d] = f2bf(val);
    }
}

// ---------------------------------------------------------------------------
extern "C" void kernel_launch(void* const* d_in, const int* in_sizes, int n_in,
                              void* d_out, int out_size, void* d_ws, size_t ws_size,
                              hipStream_t stream)
{
  (void)in_sizes; (void)n_in; (void)out_size; (void)ws_size;
  const float* x        = (const float*)d_in[0];
  const float* y        = (const float*)d_in[1];
  const int*   mask_x   = (const int*)d_in[2];
  const int*   mask_y   = (const int*)d_in[3];
  const int*   attn_m   = (const int*)d_in[4];
  const float* Wq       = (const float*)d_in[5];
  const float* bq       = (const float*)d_in[6];
  const float* Wk       = (const float*)d_in[7];
  const float* bk       = (const float*)d_in[8];
  const float* Wv       = (const float*)d_in[9];
  const float* bv       = (const float*)d_in[10];
  const float* Wo       = (const float*)d_in[11];
  const float* bo       = (const float*)d_in[12];

  // workspace: bf16 activations (20 MB total)
  unsigned short* qb   = (unsigned short*)d_ws;                         // 4096x1024
  unsigned short* kbuf = qb   + (size_t)B_ * S_ * H_  * D_;             // 4096x256
  unsigned short* vbuf = kbuf + (size_t)B_ * S_ * HK_ * D_;             // 4096x256
  unsigned short* hb   = vbuf + (size_t)B_ * S_ * HK_ * D_;             // 4096x1024

  const int M = B_ * S_;
  dim3 blk(256);
  // Q/K/V projections (fp32 in -> bf16 out, WMMA bf16, fp32 accum)
  gemm_bf16_wmma<false, true><<<dim3(DIM_ / 64,       M / 64), blk, 0, stream>>>(
      x, Wq, bq, qb, M, H_ * D_, DIM_);
  gemm_bf16_wmma<false, true><<<dim3((HK_ * D_) / 64, M / 64), blk, 0, stream>>>(
      y, Wk, bk, kbuf, M, HK_ * D_, DIM_);
  gemm_bf16_wmma<false, true><<<dim3((HK_ * D_) / 64, M / 64), blk, 0, stream>>>(
      y, Wv, bv, vbuf, M, HK_ * D_, DIM_);

  // fused flash attention
  attn_wmma<<<dim3(S_ / 64, H_, B_), dim3(128), 0, stream>>>(
      qb, kbuf, vbuf, mask_x, mask_y, attn_m, hb);

  // output projection (bf16 in -> fp32 out)
  gemm_bf16_wmma<true, false><<<dim3(DIM_ / 64, M / 64), blk, 0, stream>>>(
      hb, Wo, bo, d_out, M, DIM_, DIM_);
}
